// VectorQuantizer_58256936403108
// MI455X (gfx1250) — matrix operand order repaired and checked by reading, not compiled, not run on hardware
//
#include <hip/hip_runtime.h>
#include <hip/hip_bf16.h>

typedef __attribute__((ext_vector_type(16))) _Float16 v16h;
typedef __attribute__((ext_vector_type(8)))  _Float16 v8h;
typedef __attribute__((ext_vector_type(8)))  float    v8f;
typedef __attribute__((ext_vector_type(4)))  unsigned u32x4;
typedef __attribute__((ext_vector_type(8)))  int      i32x8;
typedef __attribute__((ext_vector_type(4)))  int      i32x4;
typedef __attribute__((address_space(3)))    _Float16 lds_h;

#define N_TOK 32768   // 32 * 32 * 32 rows of zf
#define N_EMB 8192
#define EDIM  256
#define HW    1024    // 32*32 spatial per image
#define NCHUNK 128    // 8192 / 64 columns per chunk
#define EB_STRIDE 264 // 256 + 8 halves pad -> conflict-free b128 LDS reads

// ---------------------------------------------------------------------------
// Kernel 1: z [32,256,32,32] f32 -> zf [N_TOK,256] f16  (+ exact f32 norms)
// ---------------------------------------------------------------------------
__global__ __launch_bounds__(256) void pack_z_kernel(
    const float* __restrict__ z, _Float16* __restrict__ zf,
    float* __restrict__ znorm) {
  __shared__ float red[256];
  const int n  = blockIdx.x;
  const int c  = threadIdx.x;
  const int b  = n >> 10;
  const int hw = n & 1023;
  float v = z[((size_t)b * EDIM + c) * HW + hw];
  zf[(size_t)n * EDIM + c] = (_Float16)v;
  red[c] = v * v;
  __syncthreads();
  for (int s = 128; s > 0; s >>= 1) {
    if (c < s) red[c] += red[c + s];
    __syncthreads();
  }
  if (c == 0) znorm[n] = red[0];
}

// ---------------------------------------------------------------------------
// Kernel 2: embedding [N_EMB,256] f32 -> ef f16 (+ exact f32 norms)
// ---------------------------------------------------------------------------
__global__ __launch_bounds__(256) void pack_e_kernel(
    const float* __restrict__ emb, _Float16* __restrict__ ef,
    float* __restrict__ enorm) {
  __shared__ float red[256];
  const int e = blockIdx.x;
  const int c = threadIdx.x;
  float v = emb[(size_t)e * EDIM + c];
  ef[(size_t)e * EDIM + c] = (_Float16)v;
  red[c] = v * v;
  __syncthreads();
  for (int s = 128; s > 0; s >>= 1) {
    if (c < s) red[c] += red[c + s];
    __syncthreads();
  }
  if (c == 0) enorm[e] = red[0];
}

// ---------------------------------------------------------------------------
// TDM descriptor issue: DMA one 64-row x 256-half tile of ef into LDS with
// +8-half row padding (pad_interval code 6 = 128 dwords, pad_amount code 3 =
// 4 dwords), data_size=2B. D# layout per CDNA5 ISA.
// ---------------------------------------------------------------------------
__device__ __forceinline__ void tdm_load_chunk(const _Float16* gsrc,
                                               unsigned lds_off) {
  unsigned long long ga = (unsigned long long)(uintptr_t)gsrc;
  u32x4 g0;
  g0.x = 1u;                                   // count=1 (valid user D#)
  g0.y = lds_off;                              // lds_addr (bytes)
  g0.z = (unsigned)(ga & 0xffffffffu);         // global_addr[31:0]
  g0.w = (unsigned)((ga >> 32) & 0x1ffffffu)   // global_addr[56:32]
         | (2u << 30);                         // type = 2 ("image")
  i32x8 g1;
  g1[0] = (1 << 16)      // data_size = 1 -> 2 bytes
        | (1 << 20)      // pad_enable
        | (6 << 22)      // pad_interval: 128 dwords (one 512B tile row)
        | (3 << 25);     // pad_amount: 4 dwords (16B -> 264-half stride)
  g1[1] = (int)(256u << 16);   // tensor_dim0[15:0] = 256 halves
  g1[2] = (int)(64u << 16);    // tensor_dim1[15:0] = 64 rows
  g1[3] = (int)(256u << 16);   // tile_dim0 = 256 halves
  g1[4] = 64;                  // tile_dim1 = 64 rows (tile_dim2 = 0)
  g1[5] = 256;                 // tensor_dim0_stride = 256 halves
  g1[6] = 0;
  g1[7] = 0;
  i32x4 gz4 = {0, 0, 0, 0};          // 2-D tile: groups 2/3 unused
  i32x8 gz8 = {0, 0, 0, 0, 0, 0, 0, 0};
  __builtin_amdgcn_tensor_load_to_lds(g0, g1, gz4, gz4, gz8, 0);
}

__device__ __forceinline__ v16h load_a_frag(const _Float16* p) {
  v8h lo = *(const v8h*)(p);
  v8h hi = *(const v8h*)(p + 16);
  return __builtin_shufflevector(lo, hi, 0, 1, 2, 3, 4, 5, 6, 7,
                                 8, 9, 10, 11, 12, 13, 14, 15);
}
__device__ __forceinline__ v16h load_b_frag(const _Float16* p) {
  v8h lo = *(const v8h*)(p);
  v8h hi = *(const v8h*)(p + 16);
  return __builtin_shufflevector(lo, hi, 0, 1, 2, 3, 4, 5, 6, 7,
                                 8, 9, 10, 11, 12, 13, 14, 15);
}

// ---------------------------------------------------------------------------
// Kernel 3: WMMA GEMM  d[n,e] = zn[n] + en[e] - 2 * dot(zf[n], ef[e])
// WG = 64 rows x 8192 cols (64-col chunks, TDM double-buffered B in LDS).
// 8 waves: w>>2 = row half (32 rows), w&3 = col tile (16 cols).
// Each wave: 2 accumulators sharing one B fragment; fragment loads are
// software-pipelined one k-step ahead so the WMMA never drains DScnt to 0.
// ---------------------------------------------------------------------------
__global__ __launch_bounds__(256) void vq_gemm_kernel(
    const _Float16* __restrict__ zf, const _Float16* __restrict__ ef,
    const float* __restrict__ znorm, const float* __restrict__ enorm,
    float* __restrict__ dmat, float* __restrict__ idxf,
    int* __restrict__ idxi) {
  __shared__ _Float16 za[64][EB_STRIDE];        // 33792 B
  __shared__ _Float16 eb[2][64][EB_STRIDE];     // 67584 B (double buffer)
  __shared__ float    minv[64][4];
  __shared__ int      mini[64][4];

  const int tid     = threadIdx.x;
  const int lane    = tid & 31;
  const int w       = tid >> 5;
  const int rhalf   = w >> 2;            // 0..1 : rows rhalf*32 .. +31
  const int ct      = w & 3;             // 0..3 : col tile of 16
  const int rowBase = blockIdx.x * 64;
  const int hiHalf  = (lane >> 4) & 1;
  const int lcol    = lane & 15;

  // LDS byte offsets of the two B buffers (for the TDM descriptor)
  const unsigned eb_off0 =
      (unsigned)(unsigned long long)(lds_h*)&eb[0][0][0];
  const unsigned eb_off1 =
      (unsigned)(unsigned long long)(lds_h*)&eb[1][0][0];

  // ---- stage A tile (64 x 256 f16) once ----
  {
    const v8h* src = (const v8h*)(zf + (size_t)rowBase * EDIM);
    #pragma unroll
    for (int i = 0; i < 8; ++i) {
      int v  = tid + 256 * i;            // 0..2047 (8-half vectors)
      int r  = v >> 5;
      int kv = v & 31;
      *(v8h*)&za[r][kv * 8] = src[v];
    }
  }

  // kick TDM for chunk 0 into buffer 0 (one wave programs the DMA)
  if (w == 0) tdm_load_chunk(ef, eb_off0);

  // per-lane rows (C layout: lanes 0-15 hold M=j, lanes 16-31 hold M=j+8)
  int   rl0[8], rl1[8];
  float zn0[8], zn1[8];
  #pragma unroll
  for (int j = 0; j < 8; ++j) {
    rl0[j] = rhalf * 32 + (hiHalf ? (j + 8) : j);
    rl1[j] = rl0[j] + 16;
    zn0[j] = znorm[rowBase + rl0[j]];
    zn1[j] = znorm[rowBase + rl1[j]];
  }

  float bestv0[8], bestv1[8];
  int   besti0[8], besti1[8];
  #pragma unroll
  for (int j = 0; j < 8; ++j) {
    bestv0[j] = 3.4e38f; besti0[j] = 0;
    bestv1[j] = 3.4e38f; besti1[j] = 0;
  }

  const int kbA = hiHalf ? 8 : 0;   // A layout: K halves 0-7 / 8-15 per half
  const int kbB = hiHalf ? 8 : 0;   // B layout: K halves 0-7 / 8-15 per half
  const int am0 = rhalf * 32 + lcol;
  const int am1 = am0 + 16;
  const int bn  = ct * 16 + lcol;

  for (int cc = 0; cc < NCHUNK; ++cc) {
    // pipeline: start DMA of next chunk, then wait for current chunk only
    if (w == 0) {
      if (cc + 1 < NCHUNK) {
        tdm_load_chunk(ef + (size_t)(cc + 1) * 64 * EDIM,
                       ((cc + 1) & 1) ? eb_off1 : eb_off0);
        __builtin_amdgcn_s_wait_tensorcnt((short)1);   // current chunk landed
      } else {
        __builtin_amdgcn_s_wait_tensorcnt((short)0);
      }
    }
    __syncthreads();   // B[cc&1] visible to all waves

    const _Float16* ebuf = &eb[cc & 1][0][0];
    v8f acc0 = {0.f, 0.f, 0.f, 0.f, 0.f, 0.f, 0.f, 0.f};
    v8f acc1 = {0.f, 0.f, 0.f, 0.f, 0.f, 0.f, 0.f, 0.f};

    // software-pipelined fragment loads: k+1 issued before k is consumed
    v16h a0c = load_a_frag(&za[am0][kbA]);
    v16h a1c = load_a_frag(&za[am1][kbA]);
    v16h bc  = load_b_frag(ebuf + bn * EB_STRIDE + kbB);
    #pragma unroll
    for (int kk = 0; kk < 8; ++kk) {
      v16h a0n, a1n, bnn;
      if (kk < 7) {
        a0n = load_a_frag(&za[am0][(kk + 1) * 32 + kbA]);
        a1n = load_a_frag(&za[am1][(kk + 1) * 32 + kbA]);
        bnn = load_b_frag(ebuf + bn * EB_STRIDE + (kk + 1) * 32 + kbB);
      }
      acc0 = __builtin_amdgcn_wmma_f32_16x16x32_f16(
          false, a0c, false, bc, (short)0, acc0, false, false);
      acc1 = __builtin_amdgcn_wmma_f32_16x16x32_f16(
          false, a1c, false, bc, (short)0, acc1, false, false);
      if (kk < 7) { a0c = a0n; a1c = a1n; bc = bnn; }
    }

    // ---- fused epilogue: d = zn + en - 2*acc ; NT stream out + argmin ----
    const int colAbs = cc * 64 + ct * 16 + lcol;
    const float en = enorm[colAbs];
    #pragma unroll
    for (int j = 0; j < 8; ++j) {
      float v0 = zn0[j] + en - 2.0f * acc0[j];
      float v1 = zn1[j] + en - 2.0f * acc1[j];
      __builtin_nontemporal_store(
          v0, &dmat[(size_t)(rowBase + rl0[j]) * N_EMB + colAbs]);
      __builtin_nontemporal_store(
          v1, &dmat[(size_t)(rowBase + rl1[j]) * N_EMB + colAbs]);
      if (v0 < bestv0[j]) { bestv0[j] = v0; besti0[j] = colAbs; }
      if (v1 < bestv1[j]) { bestv1[j] = v1; besti1[j] = colAbs; }
    }
    __syncthreads();   // all reads of B[cc&1] done before TDM refills it
  }

  // ---- argmin: reduce across the 16 lanes holding each row ----
  #pragma unroll
  for (int s = 1; s < 16; s <<= 1) {
    #pragma unroll
    for (int j = 0; j < 8; ++j) {
      float ov; int oi;
      ov = __shfl_xor(bestv0[j], s, 32);
      oi = __shfl_xor(besti0[j], s, 32);
      if (ov < bestv0[j] || (ov == bestv0[j] && oi < besti0[j])) {
        bestv0[j] = ov; besti0[j] = oi;
      }
      ov = __shfl_xor(bestv1[j], s, 32);
      oi = __shfl_xor(besti1[j], s, 32);
      if (ov < bestv1[j] || (ov == bestv1[j] && oi < besti1[j])) {
        bestv1[j] = ov; besti1[j] = oi;
      }
    }
  }
  if (lcol == 0) {   // lanes 0 and 16 hold the group minima
    #pragma unroll
    for (int j = 0; j < 8; ++j) {
      minv[rl0[j]][ct] = bestv0[j]; mini[rl0[j]][ct] = besti0[j];
      minv[rl1[j]][ct] = bestv1[j]; mini[rl1[j]][ct] = besti1[j];
    }
  }
  __syncthreads();
  if (tid < 64) {
    float bv = minv[tid][0];
    int   bi = mini[tid][0];
    #pragma unroll
    for (int q = 1; q < 4; ++q) {
      float v2 = minv[tid][q];
      int   i2 = mini[tid][q];
      if (v2 < bv || (v2 == bv && i2 < bi)) { bv = v2; bi = i2; }
    }
    const int n = rowBase + tid;
    idxi[n] = bi;
    idxf[n] = (float)bi;
  }
}

// ---------------------------------------------------------------------------
// Kernel 4: gather z_q = z + (E[idx] - z)  (straight-through arithmetic),
// write back in b,c,h,w layout, per-block partial sum of (z_q - z)^2
// ---------------------------------------------------------------------------
__global__ __launch_bounds__(256) void gather_kernel(
    const float* __restrict__ z, const float* __restrict__ emb,
    const int* __restrict__ idxi, float* __restrict__ zq,
    float* __restrict__ partial) {
  __shared__ float red[256];
  const int n  = blockIdx.x;
  const int c  = threadIdx.x;
  const int b  = n >> 10;
  const int hw = n & 1023;
  const int e  = idxi[n];
  const size_t zi = ((size_t)b * EDIM + c) * HW + hw;
  float zv = z[zi];
  float qv = emb[(size_t)e * EDIM + c];
  float diff = qv - zv;
  __builtin_nontemporal_store(zv + diff, &zq[zi]);  // == reference z+sg(zq-z)
  red[c] = diff * diff;
  __syncthreads();
  for (int s = 128; s > 0; s >>= 1) {
    if (c < s) red[c] += red[c + s];
    __syncthreads();
  }
  if (c == 0) partial[n] = red[0];
}

// ---------------------------------------------------------------------------
// Kernel 5: deterministic final loss reduction: (1+beta) * mean(diff^2)
// ---------------------------------------------------------------------------
__global__ __launch_bounds__(256) void loss_kernel(
    const float* __restrict__ partial, float* __restrict__ loss) {
  __shared__ float red[256];
  float s = 0.f;
  for (int i = threadIdx.x; i < N_TOK; i += 256) s += partial[i];
  red[threadIdx.x] = s;
  __syncthreads();
  for (int t = 128; t > 0; t >>= 1) {
    if (threadIdx.x < t) red[threadIdx.x] += red[threadIdx.x + t];
    __syncthreads();
  }
  if (threadIdx.x == 0) *loss = red[0] * 1.25f / 8388608.0f;
}

// ---------------------------------------------------------------------------
extern "C" void kernel_launch(void* const* d_in, const int* in_sizes, int n_in,
                              void* d_out, int out_size, void* d_ws,
                              size_t ws_size, hipStream_t stream) {
  const float* z   = (const float*)d_in[0];   // [32,256,32,32]
  const float* emb = (const float*)d_in[1];   // [8192,256]

  // workspace layout (bytes)
  char* ws = (char*)d_ws;
  _Float16* zf      = (_Float16*)(ws);                            // 16 MB
  _Float16* ef      = (_Float16*)(ws + (16u << 20));              // 4 MB
  float*    znorm   = (float*)(ws + (20u << 20));                 // 128 KB
  float*    enorm   = (float*)(ws + (20u << 20) + (128u << 10));  // 32 KB
  int*      idxi    = (int*)(ws + (20u << 20) + (160u << 10));    // 128 KB
  float*    partial = (float*)(ws + (20u << 20) + (288u << 10));  // 128 KB

  // output layout: [z_q (8388608)] [loss (1)] [d (268435456)] [indices (32768)]
  float* out  = (float*)d_out;
  float* zq   = out;
  float* loss = out + 8388608;
  float* dmat = out + 8388609;
  float* idxf = out + 8388609 + (size_t)N_TOK * N_EMB;

  pack_z_kernel<<<N_TOK, 256, 0, stream>>>(z, zf, znorm);
  pack_e_kernel<<<N_EMB, 256, 0, stream>>>(emb, ef, enorm);
  vq_gemm_kernel<<<N_TOK / 64, 256, 0, stream>>>(zf, ef, znorm, enorm,
                                                 dmat, idxf, idxi);
  gather_kernel<<<N_TOK, 256, 0, stream>>>(z, emb, idxi, zq, partial);
  loss_kernel<<<1, 256, 0, stream>>>(partial, loss);
}
